// TransitionDown_31817117728962
// MI455X (gfx1250) — matrix operand order repaired and checked by reading, not compiled, not run on hardware
//
#include <hip/hip_runtime.h>
#include <hip/hip_bf16.h>

typedef __attribute__((ext_vector_type(16))) _Float16 v16h;
typedef __attribute__((ext_vector_type(8)))  _Float16 v8h;
typedef __attribute__((ext_vector_type(8)))  float    v8f;

#define K_NB   16   // neighbors per query (WMMA M)
#define C_IN   32
#define C_OUT  64
#define DIN    35   // 3 + C_IN
#define KPAD   64   // K padded to 2x WMMA K=32
#define WAVES_PER_BLOCK 8
#define BLOCK  (WAVES_PER_BLOCK * 32)

// A-tile K permutation: ISA 16-bit A 16x32 fragment wants K groups
// {0-7,16-23} for lanes 0-15 and {8-15,24-31} for lanes 16-31 (per 32-block).
// Storing column k at position p(k) = swap(bit3,bit4 of k) makes each lane's
// 16 fragment halves a contiguous 32-byte run -> v16h load = 2x ds_load_b128.

__global__ __launch_bounds__(BLOCK)
void td_wmma_kernel(const float* __restrict__ point,
                    const float* __restrict__ feat,
                    const int*   __restrict__ idx,
                    const int*   __restrict__ sample_idx,
                    const float* __restrict__ W,
                    const float* __restrict__ gamma,
                    const float* __restrict__ beta,
                    const float* __restrict__ rmean,
                    const float* __restrict__ rvar,
                    float* __restrict__ out,
                    int nq)
{
    __shared__ __attribute__((aligned(32))) _Float16 Wt[C_OUT * KPAD];  // [c][k], k>=35 zero
    __shared__ float s_scale[C_OUT];
    __shared__ float s_shift[C_OUT];
    __shared__ __attribute__((aligned(32))) _Float16 Atile[WAVES_PER_BLOCK][K_NB * KPAD];

    const int tid  = threadIdx.x;
    const int wave = tid >> 5;
    const int lane = tid & 31;

    // ---- stage W transposed as f16 in LDS: each thread fills 16 contiguous slots of the permuted K order
    {
        const int c  = tid >> 2;        // 0..63 output channel
        const int k0 = (tid & 3) * 16;  // 0,16,32,48
        v8h w0 = {}, w1 = {};
        #pragma unroll
        for (int e = 0; e < 8; ++e) {
            int ka = (k0 & 32) + ((k0 & 16) >> 1) + e, kb = ka + 16;
            if (ka < DIN) w0[e] = (_Float16)W[ka * C_OUT + c];
            if (kb < DIN) w1[e] = (_Float16)W[kb * C_OUT + c];
        }
        *(v8h*)(&Wt[c * KPAD + k0])     = w0;
        *(v8h*)(&Wt[c * KPAD + k0 + 8]) = w1;
    }
    // ---- fold BN running stats into per-channel scale/shift ----
    if (tid < C_OUT) {
        float inv = rsqrtf(rvar[tid] + 1e-5f);
        float sc  = gamma[tid] * inv;
        s_scale[tid] = sc;
        s_shift[tid] = beta[tid] - rmean[tid] * sc;
    }

    const int  m      = blockIdx.x * WAVES_PER_BLOCK + wave;   // one query per wave
    const bool active = (m < nq);

    // ---- gather: 2 lanes per neighbor build the permuted f16 A row ----
    // Permuted chunk order (8 halves each): pos0=K0-7, pos8=K16-23, pos16=K8-15,
    // pos24=K24-31, pos32=K32-39, pos40=K48-55, pos48=K40-47, pos56=K56-63.
    if (active) {
        const int n  = lane & 15;   // neighbor row (WMMA M)
        const int hf = lane >> 4;
        const int j  = idx[m * K_NB + n];
        const float* fj = feat + (size_t)j * C_IN;
        _Float16* arow = &Atile[wave][n * KPAD];
        if (hf == 0) {
            // K 0..15: centered xyz + feat[0..12]
            const int s = sample_idx[m];
            v8h c0, c2;
            c0[0] = (_Float16)(point[j * 3 + 0] - point[s * 3 + 0]);
            c0[1] = (_Float16)(point[j * 3 + 1] - point[s * 3 + 1]);
            c0[2] = (_Float16)(point[j * 3 + 2] - point[s * 3 + 2]);
            #pragma unroll
            for (int e = 0; e < 5; ++e) c0[3 + e] = (_Float16)fj[e];      // f0..f4
            #pragma unroll
            for (int e = 0; e < 8; ++e) c2[e] = (_Float16)fj[5 + e];      // f5..f12 (K8-15)
            *(v8h*)(arow + 0)  = c0;   // K 0..7
            *(v8h*)(arow + 16) = c2;   // K 8..15 lives at pos 16
        } else {
            // K 16..39: feat[13..31], then zero K 40..63
            v8h c1, c3, c4, z = {};
            #pragma unroll
            for (int e = 0; e < 8; ++e) c1[e] = (_Float16)fj[13 + e];     // f13..f20 (K16-23)
            #pragma unroll
            for (int e = 0; e < 8; ++e) c3[e] = (_Float16)fj[21 + e];     // f21..f28 (K24-31)
            c4 = z;
            c4[0] = (_Float16)fj[29];                                     // f29..f31 (K32-34)
            c4[1] = (_Float16)fj[30];
            c4[2] = (_Float16)fj[31];
            *(v8h*)(arow + 8)  = c1;   // K 16..23 at pos 8
            *(v8h*)(arow + 24) = c3;   // K 24..31
            *(v8h*)(arow + 32) = c4;   // K 32..39
            *(v8h*)(arow + 40) = z;    // K 48..55
            *(v8h*)(arow + 48) = z;    // K 40..47
            *(v8h*)(arow + 56) = z;    // K 56..63
        }
    }
    __syncthreads();

    if (active) {
        const int row = lane & 15;
        const int hi  = lane >> 4;

        // A fragments: contiguous thanks to permuted tile layout
        const _Float16* ar = &Atile[wave][row * KPAD];
        v16h a0 = *(const v16h*)(ar + 16 * hi);        // K 0..31 fragment
        v16h a1 = *(const v16h*)(ar + 32 + 16 * hi);   // K 32..63 fragment

        // B fragments: Wt shares the A tile's permuted K layout, so per lane:
        //   elem e -> K = kb*32 + 8*hi + e (e<8), kb*32 + 16 + 8*hi + (e-8) (e>=8)
        v16h bf[4][2];
        #pragma unroll
        for (int nt = 0; nt < 4; ++nt) {
            const _Float16* bcol = &Wt[(nt * 16 + (lane & 15)) * KPAD];
            bf[nt][0] = *(const v16h*)(bcol + 16 * hi);
            bf[nt][1] = *(const v16h*)(bcol + 32 + 16 * hi);
        }

        // D = A(16x64) x W(64x64): 4 independent chains, dependent pairs 4 apart
        v8f acc[4];
        #pragma unroll
        for (int nt = 0; nt < 4; ++nt) {
            v8f c = {};
            acc[nt] = __builtin_amdgcn_wmma_f32_16x16x32_f16(
                false, a0, false, bf[nt][0], (short)0, c, false, false);
        }
        #pragma unroll
        for (int nt = 0; nt < 4; ++nt) {
            acc[nt] = __builtin_amdgcn_wmma_f32_16x16x32_f16(
                false, a1, false, bf[nt][1], (short)0, acc[nt], false, false);
        }

        // Epilogue: BN + ReLU + max over 16 neighbor rows.
        // C/D layout: lane holds rows M = r + 8*(lane>>4), col N = lane&15.
        #pragma unroll
        for (int nt = 0; nt < 4; ++nt) {
            const int ch = nt * 16 + (lane & 15);
            const float sc = s_scale[ch];
            const float sh = s_shift[ch];
            float mx = 0.0f;                        // ReLU floor folds into the max
            #pragma unroll
            for (int r = 0; r < 8; ++r)
                mx = fmaxf(mx, fmaf(acc[nt][r], sc, sh));
            mx = fmaxf(mx, __shfl_xor(mx, 16, 32)); // merge rows 0-7 with 8-15 (wave32)
            if (lane < 16)
                out[m * C_OUT + ch] = mx;
        }
    }
}

extern "C" void kernel_launch(void* const* d_in, const int* in_sizes, int n_in,
                              void* d_out, int out_size, void* d_ws, size_t ws_size,
                              hipStream_t stream) {
    const float* point = (const float*)d_in[0];
    const float* feat  = (const float*)d_in[1];
    const int*   idx   = (const int*)d_in[2];
    const int*   sidx  = (const int*)d_in[3];
    const float* W     = (const float*)d_in[4];
    const float* gamma = (const float*)d_in[5];
    const float* beta  = (const float*)d_in[6];
    const float* rmean = (const float*)d_in[7];
    const float* rvar  = (const float*)d_in[8];
    float* out = (float*)d_out;

    const int nq = in_sizes[3];                                  // N_QUERIES
    const int blocks = (nq + WAVES_PER_BLOCK - 1) / WAVES_PER_BLOCK;
    td_wmma_kernel<<<blocks, BLOCK, 0, stream>>>(point, feat, idx, sidx, W,
                                                 gamma, beta, rmean, rvar, out, nq);
}
